// SentenceTaggerRNN_42296837931711
// MI455X (gfx1250) — matrix operand order repaired and checked by reading, not compiled, not run on hardware
//
#include <hip/hip_runtime.h>
#include <hip/hip_bf16.h>
#include <math.h>

typedef __attribute__((ext_vector_type(16))) _Float16 v16h;
typedef __attribute__((ext_vector_type(8)))  float    v8f;
typedef int ai4 __attribute__((vector_size(16)));  // matches builtin param pointee

#define HD 128   // hidden size H
#define GD 512   // 4*H gate width
#define DD 256   // 2*H bidirectional width

__device__ __forceinline__ float sig_f(float x) { return 1.0f / (1.0f + expf(-x)); }

__device__ __forceinline__ v8f wmma16x16x32(v16h a, v16h b, v8f c) {
  // D = A(16x32 f16) * B(32x16 f16) + C(16x16 f32)
  return __builtin_amdgcn_wmma_f32_16x16x32_f16(false, a, false, b, (short)0, c,
                                                false, false);
}

// ---- CDNA5 async global->LDS copy (ASYNCcnt-tracked), with fallback ---------
#if defined(__has_builtin)
#if __has_builtin(__builtin_amdgcn_global_load_async_to_lds_b128)
#define ATHENA_ASYNC_LDS 1
#endif
#endif

__device__ __forceinline__ void async_copy_b128(void* lds_dst, const void* gsrc) {
#if defined(ATHENA_ASYNC_LDS)
  __builtin_amdgcn_global_load_async_to_lds_b128(
      (__attribute__((address_space(1))) ai4*)(void*)gsrc,
      (__attribute__((address_space(3))) ai4*)lds_dst, 0, 0);
#else
  *(uint4*)lds_dst = *(const uint4*)gsrc;
#endif
}

__device__ __forceinline__ void async_wait_all() {
#if defined(ATHENA_ASYNC_LDS)
#if __has_builtin(__builtin_amdgcn_s_wait_asynccnt)
  __builtin_amdgcn_s_wait_asynccnt(0);
#else
  asm volatile("s_wait_asynccnt 0x0" ::: "memory");
#endif
#endif
}

// cooperative tile stage: nBytes must be a multiple of 16; 256 threads
__device__ __forceinline__ void stage_to_lds_async(void* lds, const void* gsrc,
                                                   int nBytes, int tid) {
  char* d = (char*)lds;
  const char* s = (const char*)gsrc;
  for (int i = tid; i < (nBytes >> 4); i += 256)
    async_copy_b128(d + ((size_t)i << 4), s + ((size_t)i << 4));
}

// ---- fragment loaders matching the CDNA5 16-bit WMMA VGPR layouts -----------
// A 16x32 (MxK): lane<16 holds row m=lane, K = {8h..8h+7, 16+8h..16+8h+7}, h=lane>=16
__device__ __forceinline__ v16h load_a_f32(const float* __restrict__ base, size_t ld,
                                           int lane) {
  const int hf = (lane >> 4) & 1;
  const int m  = lane & 15;
  const float* p = base + (size_t)m * ld + 8 * hf;
  v16h a;
#pragma unroll
  for (int j = 0; j < 8; ++j) a[j] = (_Float16)p[j];
#pragma unroll
  for (int j = 0; j < 8; ++j) a[8 + j] = (_Float16)p[16 + j];
  return a;
}

__device__ __forceinline__ v16h load_a_f16(const _Float16* base, int ld, int lane) {
  const int hf = (lane >> 4) & 1;
  const int m  = lane & 15;
  const _Float16* p = base + m * ld + 8 * hf;
  v16h a;
#pragma unroll
  for (int j = 0; j < 8; ++j) a[j] = p[j];
#pragma unroll
  for (int j = 0; j < 8; ++j) a[8 + j] = p[16 + j];
  return a;
}

// B 32x16 (KxN): B[k][n] = W[n0+n][k0+k]; K = {8h..8h+7, 16+8h..16+8h+7}, h=lane>=16
__device__ __forceinline__ v16h load_b_f16(const _Float16* w, int ldw, int n0, int k0,
                                           int lane) {
  const int hf = (lane >> 4) & 1;
  const int n  = lane & 15;
  const _Float16* p = w + (size_t)(n0 + n) * ldw + k0 + 8 * hf;
  v16h b;
#pragma unroll
  for (int j = 0; j < 8; ++j) b[j] = p[j];
#pragma unroll
  for (int j = 0; j < 8; ++j) b[8 + j] = p[16 + j];
  return b;
}

// ---- weight f32 -> f16 ------------------------------------------------------
__global__ void f32_to_f16_kernel(const float* __restrict__ s, _Float16* __restrict__ d,
                                  int n) {
  int i = blockIdx.x * blockDim.x + threadIdx.x;
  if (i < n) d[i] = (_Float16)s[i];
}

// ---- embedding gather -------------------------------------------------------
__global__ void gather_kernel(const int* __restrict__ ids, const float* __restrict__ emb,
                              float* __restrict__ tok, int n) {
  int t = blockIdx.x;
  if (t >= n) return;
  tok[(size_t)t * 128 + threadIdx.x] = emb[(size_t)ids[t] * 128 + threadIdx.x];
}

// ---- pre-gate GEMM: pre(M,512) = X(M,E) @ W(512,E)^T + bias -----------------
// grid = (ceil(M/128), 2 dirs) x 256 threads; each wave owns a 16-row M tile.
// W (512xE f16) is staged into LDS via async global->LDS (128/256 KB, CDNA5 WGP LDS).
template <int E>
__global__ __launch_bounds__(256) void pregate_kernel(
    const float* __restrict__ X, const _Float16* __restrict__ W,
    const float* __restrict__ bias, float* __restrict__ pre, int M, size_t wStride,
    size_t preStride) {
  extern __shared__ char psm[];
  _Float16* sW = (_Float16*)psm;  // 512*E halfs
  const int tid  = threadIdx.x;
  const int lane = tid & 31;
  const int wave = tid >> 5;
  const int dir  = blockIdx.y;
  const _Float16* w = W + dir * wStride;
  const float* b    = bias + (size_t)dir * GD;
  float* p          = pre + dir * preStride;
  const int m0 = blockIdx.x * 128 + wave * 16;
  const bool active = (m0 < M);

  stage_to_lds_async(sW, w, GD * E * 2, tid);  // async copy W tile

  constexpr int NK = E / 32;
  v16h afr[NK];
  if (active) {  // overlap A-fragment global loads with the async W copy
#pragma unroll
    for (int kk = 0; kk < NK; ++kk)
      afr[kk] = load_a_f32(X + (size_t)m0 * E + kk * 32, E, lane);
  }
  async_wait_all();
  __syncthreads();
  if (!active) return;

  const int hf = (lane >> 4) & 1;
  const int nn = lane & 15;
  for (int nt = 0; nt < GD / 16; ++nt) {
    const int nb = nt * 16;
    v8f c = {};
#pragma unroll
    for (int kk = 0; kk < NK; ++kk)
      c = wmma16x16x32(afr[kk], load_b_f16(sW, E, nb, kk * 32, lane), c);
    const float bv = b[nb + nn];
    float* o = p + (size_t)m0 * GD + nb;
#pragma unroll
    for (int r = 0; r < 8; ++r) o[(size_t)(r + 8 * hf) * GD + nn] = c[r] + bv;
  }
}

// ---- LSTM recurrence: per step g = pre[:,t,:] + h @ Whh^T; gate math --------
// grid = (NR/16, 2 dirs) x 256 threads. W_hh async-staged into LDS (128 KB).
__global__ __launch_bounds__(256) void lstm_rec_kernel(
    const float* __restrict__ pre, const _Float16* __restrict__ whh,
    float* __restrict__ out, int T, size_t preStride, size_t whhStride) {
  extern __shared__ char smem[];
  _Float16* sW   = (_Float16*)smem;                       // 512*128 f16 = 128 KB
  float*    gbuf = (float*)(smem + GD * HD * 2);          // 16*512 f32  =  32 KB
  float*    cbuf = gbuf + 16 * GD;                        // 16*128 f32  =   8 KB
  _Float16* hbuf = (_Float16*)(cbuf + 16 * HD);           // 16*128 f16  =   4 KB
  const int tid = threadIdx.x, lane = tid & 31, wave = tid >> 5;
  const int dir = blockIdx.y;
  const float* preD  = pre + dir * preStride;
  const _Float16* wD = whh + dir * whhStride;
  const int m0 = blockIdx.x * 16;

  stage_to_lds_async(sW, wD, GD * HD * 2, tid);  // async W_hh copy ...
  for (int i = tid; i < 16 * HD; i += 256) {     // ... overlapped with state init
    cbuf[i] = 0.0f;
    hbuf[i] = (_Float16)0.0f;
  }
  async_wait_all();
  __syncthreads();

  const int hf = (lane >> 4) & 1;
  const int nn = lane & 15;
  for (int t = 0; t < T; ++t) {
    const int ts = dir ? (T - 1 - t) : t;
    if (t + 1 < T) {
      const int tn = dir ? (ts - 1) : (ts + 1);
      __builtin_prefetch(preD + ((size_t)m0 * T + tn) * GD, 0, 0);
    }
    v16h afr[4];
#pragma unroll
    for (int kk = 0; kk < 4; ++kk) afr[kk] = load_a_f16(hbuf + kk * 32, HD, lane);
#pragma unroll
    for (int ntl = 0; ntl < 4; ++ntl) {
      const int nb = (wave * 4 + ntl) * 16;
      const float* cs = preD + ((size_t)m0 * T + ts) * GD + nb;
      v8f c;
#pragma unroll
      for (int r = 0; r < 8; ++r) c[r] = cs[(size_t)(r + 8 * hf) * T * GD + nn];
#pragma unroll
      for (int kk = 0; kk < 4; ++kk)
        c = wmma16x16x32(afr[kk], load_b_f16(sW, HD, nb, kk * 32, lane), c);
#pragma unroll
      for (int r = 0; r < 8; ++r) gbuf[(size_t)(r + 8 * hf) * GD + nb + nn] = c[r];
    }
    __syncthreads();
    for (int e = tid; e < 16 * HD; e += 256) {
      const int m = e >> 7, j = e & 127;
      const float* g = gbuf + (size_t)m * GD;
      const float ig = sig_f(g[j]);
      const float fg = sig_f(g[HD + j]);
      const float gg = tanhf(g[2 * HD + j]);
      const float og = sig_f(g[3 * HD + j]);
      const float cc = fg * cbuf[e] + ig * gg;
      const float hh = og * tanhf(cc);
      cbuf[e] = cc;
      hbuf[e] = (_Float16)hh;
      out[((size_t)(m0 + m) * T + ts) * DD + dir * HD + j] = hh;
    }
    __syncthreads();
  }
}

// ---- mean over time axis: in(R,T,D) -> out(R,D) -----------------------------
__global__ void mean_time_kernel(const float* __restrict__ in, float* __restrict__ outp,
                                 int T, int D) {
  const int r = blockIdx.x, d = threadIdx.x;
  const float* p = in + (size_t)r * T * D + d;
  float s = 0.0f;
  for (int t = 0; t < T; ++t) s += p[(size_t)t * D];
  outp[(size_t)r * D + d] = s * (1.0f / (float)T);
}

// ---- y(N,O) = [tanh](x(N,D) @ w(O,D)^T + b) ---------------------------------
__global__ void linear_kernel(const float* __restrict__ x, const float* __restrict__ w,
                              const float* __restrict__ b, float* __restrict__ y, int N,
                              int D, int O, int do_tanh) {
  int idx = blockIdx.x * blockDim.x + threadIdx.x;
  if (idx >= N * O) return;
  const int n = idx / O, o = idx - n * O;
  float s = b[o];
  const float* xr = x + (size_t)n * D;
  const float* wr = w + (size_t)o * D;
  for (int k = 0; k < D; ++k) s += xr[k] * wr[k];
  y[idx] = do_tanh ? tanhf(s) : s;
}

// ---- pred[b,s] = out.content_w + content_b + out.sal_vec[b] -----------------
__global__ void pred_kernel(const float* __restrict__ out, const float* __restrict__ cw,
                            const float* __restrict__ cb, const float* __restrict__ salv,
                            float* __restrict__ pred, int B, int S, int D) {
  int idx = blockIdx.x * blockDim.x + threadIdx.x;
  if (idx >= B * S) return;
  const int b = idx / S;
  const float* o  = out + (size_t)idx * D;
  const float* sv = salv + (size_t)b * D;
  float c = cb[0], sl = 0.0f;
  for (int k = 0; k < D; ++k) {
    c += o[k] * cw[k];
    sl += o[k] * sv[k];
  }
  pred[idx] = c + sl;
}

// ---- novelty scan, single workgroup; summary state resident in LDS ----------
__global__ __launch_bounds__(256) void novelty_kernel(
    const float* __restrict__ out, const float* __restrict__ pred,
    const float* __restrict__ nov_w, const float* __restrict__ nov_b,
    float* __restrict__ result, int B, int S, int D) {
  extern __shared__ char nsm[];
  float* summary = (float*)nsm;        // B*D
  float* tanhS   = summary + B * D;    // B*D
  float* nvec    = tanhS + B * D;      // B*D
  __shared__ float nov_i[64];
  const int tid = threadIdx.x;
  for (int i = tid; i < B * D; i += 256) summary[i] = 0.0f;
  __syncthreads();
  for (int step = 0; step < S; ++step) {
    for (int i = tid; i < B * D; i += 256) tanhS[i] = tanhf(summary[i]);
    __syncthreads();
    for (int i = tid; i < B * D; i += 256) {
      const int b = i / D, d = i - b * D;
      float s = nov_b[d];
      const float* tr = tanhS + (size_t)b * D;
      const float* wr = nov_w + (size_t)d * D;
      for (int k = 0; k < D; ++k) s += tr[k] * wr[k];
      nvec[i] = s;
    }
    __syncthreads();
    if (step == S - 1) {
      for (int i = tid; i < B * S; i += 256) {
        const int b = i / S, s = i - b * S;
        const float* o  = out + ((size_t)b * S + s) * D;
        const float* nv = nvec + (size_t)b * D;
        float acc = 0.0f;
        for (int k = 0; k < D; ++k) acc += o[k] * nv[k];
        result[i] = pred[i] - acc;
      }
    } else {
      if (tid < B) {
        const float* o  = out + ((size_t)tid * S + step) * D;
        const float* nv = nvec + (size_t)tid * D;
        float acc = 0.0f;
        for (int k = 0; k < D; ++k) acc += o[k] * nv[k];
        nov_i[tid] = acc;
      }
      __syncthreads();
      for (int i = tid; i < B * D; i += 256) {
        const int b = i / D, d = i - b * D;
        const float prob = sig_f(pred[b * S + step] - nov_i[b]);
        summary[i] += prob * out[((size_t)b * S + step) * D + d];
      }
      __syncthreads();
    }
  }
}

extern "C" void kernel_launch(void* const* d_in, const int* in_sizes, int n_in,
                              void* d_out, int out_size, void* d_ws, size_t ws_size,
                              hipStream_t stream) {
  const int*   ids    = (const int*)d_in[0];
  const float* emb    = (const float*)d_in[1];
  const float* ew_ih0 = (const float*)d_in[2];
  const float* ew_hh0 = (const float*)d_in[3];
  const float* eb0    = (const float*)d_in[4];
  const float* ew_ih1 = (const float*)d_in[5];
  const float* ew_hh1 = (const float*)d_in[6];
  const float* eb1    = (const float*)d_in[7];
  const float* dw_ih0 = (const float*)d_in[8];
  const float* dw_hh0 = (const float*)d_in[9];
  const float* db0    = (const float*)d_in[10];
  const float* dw_ih1 = (const float*)d_in[11];
  const float* dw_hh1 = (const float*)d_in[12];
  const float* db1    = (const float*)d_in[13];
  const float* content_w = (const float*)d_in[14];
  const float* content_b = (const float*)d_in[15];
  const float* docl_w = (const float*)d_in[16];
  const float* docl_b = (const float*)d_in[17];
  const float* sal_w  = (const float*)d_in[18];
  const float* sal_b  = (const float*)d_in[19];
  const float* nov_w  = (const float*)d_in[20];
  const float* nov_b  = (const float*)d_in[21];
  float* result = (float*)d_out;
  (void)in_sizes; (void)n_in; (void)out_size; (void)ws_size;

  const int B = 32, S = 50, T = 40, E = 128, H = 128, D = 256, G = 512;
  const int NSEQ = B * S;    // 1600 word-level sequences
  const int NTOK = NSEQ * T; // 64000 tokens

  // bump allocator over d_ws (~435 MB total)
  char* ws = (char*)d_ws;
  size_t off = 0;
  auto alloc = [&](size_t bytes) -> char* {
    char* p = ws + off;
    off += (bytes + 255) & ~(size_t)255;
    return p;
  };
  float* tok   = (float*)alloc((size_t)NTOK * E * 4);
  float* pre   = (float*)alloc((size_t)2 * NTOK * G * 4);  // reused per layer
  float* enc0  = (float*)alloc((size_t)NTOK * D * 4);
  float* enc1  = (float*)alloc((size_t)NTOK * D * 4);
  float* sent  = (float*)alloc((size_t)NSEQ * D * 4);
  float* doc0  = (float*)alloc((size_t)NSEQ * D * 4);
  float* doc1  = (float*)alloc((size_t)NSEQ * D * 4);
  float* dmean = (float*)alloc((size_t)B * D * 4);
  float* demb  = (float*)alloc((size_t)B * D * 4);
  float* salv  = (float*)alloc((size_t)B * D * 4);
  float* predb = (float*)alloc((size_t)NSEQ * 4);
  _Float16* h_ew_ih0 = (_Float16*)alloc((size_t)2 * G * E * 2);
  _Float16* h_ew_hh0 = (_Float16*)alloc((size_t)2 * G * H * 2);
  _Float16* h_ew_ih1 = (_Float16*)alloc((size_t)2 * G * D * 2);
  _Float16* h_ew_hh1 = (_Float16*)alloc((size_t)2 * G * H * 2);
  _Float16* h_dw_ih0 = (_Float16*)alloc((size_t)2 * G * D * 2);
  _Float16* h_dw_hh0 = (_Float16*)alloc((size_t)2 * G * H * 2);
  _Float16* h_dw_ih1 = (_Float16*)alloc((size_t)2 * G * D * 2);
  _Float16* h_dw_hh1 = (_Float16*)alloc((size_t)2 * G * H * 2);

  auto cvt = [&](const float* s, _Float16* dptr, int n) {
    f32_to_f16_kernel<<<(n + 255) / 256, 256, 0, stream>>>(s, dptr, n);
  };
  cvt(ew_ih0, h_ew_ih0, 2 * G * E);
  cvt(ew_hh0, h_ew_hh0, 2 * G * H);
  cvt(ew_ih1, h_ew_ih1, 2 * G * D);
  cvt(ew_hh1, h_ew_hh1, 2 * G * H);
  cvt(dw_ih0, h_dw_ih0, 2 * G * D);
  cvt(dw_hh0, h_dw_hh0, 2 * G * H);
  cvt(dw_ih1, h_dw_ih1, 2 * G * D);
  cvt(dw_hh1, h_dw_hh1, 2 * G * H);

  gather_kernel<<<NTOK, 128, 0, stream>>>(ids, emb, tok, NTOK);

  const size_t recShmem =
      (size_t)G * H * 2 + (size_t)16 * G * 4 + (size_t)16 * H * 4 + (size_t)16 * H * 2;
  const size_t pgShmem128 = (size_t)G * 128 * 2;  // 128 KB
  const size_t pgShmem256 = (size_t)G * 256 * 2;  // 256 KB (CDNA5 WGP LDS)

  // encoder layer 0
  pregate_kernel<128><<<dim3((NTOK + 127) / 128, 2), 256, pgShmem128, stream>>>(
      tok, h_ew_ih0, eb0, pre, NTOK, (size_t)G * E, (size_t)NTOK * G);
  lstm_rec_kernel<<<dim3(NSEQ / 16, 2), 256, recShmem, stream>>>(
      pre, h_ew_hh0, enc0, T, (size_t)NTOK * G, (size_t)G * H);
  // encoder layer 1
  pregate_kernel<256><<<dim3((NTOK + 127) / 128, 2), 256, pgShmem256, stream>>>(
      enc0, h_ew_ih1, eb1, pre, NTOK, (size_t)G * D, (size_t)NTOK * G);
  lstm_rec_kernel<<<dim3(NSEQ / 16, 2), 256, recShmem, stream>>>(
      pre, h_ew_hh1, enc1, T, (size_t)NTOK * G, (size_t)G * H);
  // sentence embeddings (mean over T)
  mean_time_kernel<<<NSEQ, D, 0, stream>>>(enc1, sent, T, D);
  // document layer 0
  pregate_kernel<256><<<dim3((NSEQ + 127) / 128, 2), 256, pgShmem256, stream>>>(
      sent, h_dw_ih0, db0, pre, NSEQ, (size_t)G * D, (size_t)NSEQ * G);
  lstm_rec_kernel<<<dim3(B / 16, 2), 256, recShmem, stream>>>(
      pre, h_dw_hh0, doc0, S, (size_t)NSEQ * G, (size_t)G * H);
  // document layer 1
  pregate_kernel<256><<<dim3((NSEQ + 127) / 128, 2), 256, pgShmem256, stream>>>(
      doc0, h_dw_ih1, db1, pre, NSEQ, (size_t)G * D, (size_t)NSEQ * G);
  lstm_rec_kernel<<<dim3(B / 16, 2), 256, recShmem, stream>>>(
      pre, h_dw_hh1, doc1, S, (size_t)NSEQ * G, (size_t)G * H);
  // classifier head
  mean_time_kernel<<<B, D, 0, stream>>>(doc1, dmean, S, D);
  linear_kernel<<<(B * D + 255) / 256, 256, 0, stream>>>(dmean, docl_w, docl_b, demb, B,
                                                         D, D, 1);
  linear_kernel<<<(B * D + 255) / 256, 256, 0, stream>>>(demb, sal_w, sal_b, salv, B, D,
                                                         D, 0);
  pred_kernel<<<(NSEQ + 255) / 256, 256, 0, stream>>>(doc1, content_w, content_b, salv,
                                                      predb, B, S, D);
  novelty_kernel<<<1, 256, (size_t)3 * B * D * 4, stream>>>(doc1, predb, nov_w, nov_b,
                                                            result, B, S, D);
}
